// SelfAttentionLayer_80693845557601
// MI455X (gfx1250) — matrix operand order repaired and checked by reading, not compiled, not run on hardware
//
#include <hip/hip_runtime.h>

// ---------------- problem constants ----------------
#define DMODEL 1024
#define SEQ    2048
#define BATCH  4
#define NHEADS 16
#define HDIM   64

// ---------------- types ----------------
typedef __bf16 bf16_t;
typedef __attribute__((ext_vector_type(16))) __bf16 v16bf;
typedef __attribute__((ext_vector_type(8)))  __bf16 v8bf;
typedef __attribute__((ext_vector_type(4)))  __bf16 v4bf;
typedef __attribute__((ext_vector_type(8)))  float  v8f;
typedef __attribute__((ext_vector_type(4)))  unsigned u32x4;
typedef __attribute__((ext_vector_type(8)))  int      i32x8;
typedef __attribute__((ext_vector_type(4)))  int      i32x4;

// ---------------- CDNA5 TDM availability ----------------
#if defined(__has_builtin)
#  if __has_builtin(__builtin_amdgcn_tensor_load_to_lds)
#    define USE_TDM 1
#  endif
#endif
#ifndef USE_TDM
#  define USE_TDM 0
#endif

#if defined(__has_builtin) && __has_builtin(__builtin_amdgcn_s_wait_tensorcnt)
#  define WAIT_TENSORCNT(n) __builtin_amdgcn_s_wait_tensorcnt(n)
#else
#  define WAIT_TENSORCNT(n) asm volatile("s_wait_tensorcnt " #n ::: "memory")
#endif

static __device__ __forceinline__ v8f v8f_zero() {
  v8f z;
#pragma unroll
  for (int i = 0; i < 8; ++i) z[i] = 0.0f;
  return z;
}

static __device__ __forceinline__ v8f wmma_bf16(v16bf a, v16bf b, v8f c) {
  // D = A(16x32 bf16) * B(32x16 bf16) + C(16x16 f32)
  return __builtin_amdgcn_wmma_f32_16x16x32_bf16(false, a, false, b, (short)0, c,
                                                 false, false);
}

// A-operand 16x32 bf16 fragment.  Lane L (L&15 = row) holds K-chunks
// [kb + hf*16 .. +7] and [kb + hf*16 + 8 .. +7]  (hf = lane>>4).
static __device__ __forceinline__ v16bf load_a_frag(const bf16_t* rowptr, int kb, int hf) {
  const v8bf lo = *(const v8bf*)(rowptr + kb + hf * 16);
  const v8bf hi = *(const v8bf*)(rowptr + kb + hf * 16 + 8);
  return __builtin_shufflevector(lo, hi, 0, 1, 2, 3, 4, 5, 6, 7,
                                         8, 9, 10, 11, 12, 13, 14, 15);
}

#if USE_TDM
// Issue one TDM 2-D tile load: rows x cols elements (bf16, 2 B), row-major,
// source row stride `row_stride` elements.  Tracked on TENSORcnt.
static __device__ __forceinline__ void tdm_load_2d(unsigned lds_addr, const void* gptr,
                                                   unsigned tile_x, unsigned tile_y,
                                                   unsigned long long row_stride) {
  const unsigned long long ga = (unsigned long long)gptr;
  const unsigned td0 = 0x40000000u, td1 = 0x40000000u;  // huge dims: no OOB clip
  u32x4 g0 = {
      1u,                                                   // count=1 (valid user D#)
      lds_addr,                                             // LDS byte address
      (unsigned)ga,                                         // global_addr[31:0]
      ((unsigned)(ga >> 32) & 0x01FFFFFFu) | 0x80000000u    // addr[56:32] | type=2
  };
  i32x8 g1 = {
      (int)(1u << 16),                                      // data_size=1 (2 bytes)
      (int)((td0 & 0xFFFFu) << 16),                         // tensor_dim0[15:0]
      (int)((td0 >> 16) | ((td1 & 0xFFFFu) << 16)),         // dim0[31:16], dim1[15:0]
      (int)((td1 >> 16) | (tile_x << 16)),                  // dim1[31:16], tile_dim0
      (int)tile_y,                                          // tile_dim1 (tile_dim2=0)
      (int)(unsigned)(row_stride & 0xFFFFFFFFull),          // dim0_stride[31:0]
      (int)(unsigned)((row_stride >> 32) & 0xFFFFull),      // dim0_stride[47:32]
      0                                                     // dim1_stride (unused, 2D)
  };
  i32x4 gz = {0, 0, 0, 0};
#if __clang_major__ >= 23
  i32x8 gz8 = {0, 0, 0, 0, 0, 0, 0, 0};
  __builtin_amdgcn_tensor_load_to_lds(g0, g1, gz, gz, gz8, 0);
#else
  __builtin_amdgcn_tensor_load_to_lds(g0, g1, gz, gz, 0);
#endif
}
#endif  // USE_TDM

// ---------------- f32 -> bf16 conversion (vectorized x4) ----------------
__global__ void cvt_f32_bf16_kernel(const float* __restrict__ in,
                                    bf16_t* __restrict__ out, int n4) {
  int i      = blockIdx.x * blockDim.x + threadIdx.x;
  int stride = gridDim.x * blockDim.x;
  for (; i < n4; i += stride) {
    float4 v = ((const float4*)in)[i];
    v4bf o;
    o[0] = (bf16_t)v.x; o[1] = (bf16_t)v.y; o[2] = (bf16_t)v.z; o[3] = (bf16_t)v.w;
    ((v4bf*)out)[i] = o;
  }
}

// ---------------- GEMM:  C[m,e] = sum_d A[m,d] * W[e,d]  ----------------
// A: M x 1024 bf16 row-major.  W: 1024 x 1024 bf16 row-major ([e][d]).
// Wave tile 32(M) x 64(N).  mode 0: bf16 row-major out.
// mode 1: bf16 out stored per-head transposed  [b][h][d][n]  (for V).
// mode 2: f32 out + bias (final projection).
__global__ void __launch_bounds__(32) gemm_xwT_kernel(
    const bf16_t* __restrict__ A, const bf16_t* __restrict__ W,
    const float* __restrict__ bias, void* __restrict__ outp, int mode) {
  const int lane  = threadIdx.x;
  const int row16 = lane & 15;
  const int hf    = lane >> 4;
  const int row0  = blockIdx.x * 32;
  const int col0  = blockIdx.y * 64;

  v8f acc[2][4];
#pragma unroll
  for (int mt = 0; mt < 2; ++mt)
#pragma unroll
    for (int ct = 0; ct < 4; ++ct) acc[mt][ct] = v8f_zero();

  const bf16_t* arow0 = A + (size_t)(row0 + row16) * DMODEL;
  const bf16_t* arow1 = A + (size_t)(row0 + 16 + row16) * DMODEL;

  for (int k0 = 0; k0 < DMODEL; k0 += 32) {
    v16bf af0 = load_a_frag(arow0, k0, hf);
    v16bf af1 = load_a_frag(arow1, k0, hf);
    __builtin_prefetch(arow0 + k0 + 32, 0, 1);
    v16bf wf[4];
#pragma unroll
    for (int ct = 0; ct < 4; ++ct) {
      const bf16_t* wp = W + (size_t)(col0 + ct * 16 + row16) * DMODEL + k0 + hf * 16;
      wf[ct] = *(const v16bf*)wp;
      __builtin_prefetch(wp + 32, 0, 1);
    }
#pragma unroll
    for (int ct = 0; ct < 4; ++ct) {
      acc[0][ct] = wmma_bf16(af0, wf[ct], acc[0][ct]);
      acc[1][ct] = wmma_bf16(af1, wf[ct], acc[1][ct]);
    }
  }

  // C layout: VGPR r -> row (hf*8 + r), column lane&15.
#pragma unroll
  for (int mt = 0; mt < 2; ++mt) {
#pragma unroll
    for (int ct = 0; ct < 4; ++ct) {
#pragma unroll
      for (int r = 0; r < 8; ++r) {
        const int grow = row0 + mt * 16 + hf * 8 + r;
        const int gcol = col0 + ct * 16 + row16;
        const float v  = acc[mt][ct][r];
        if (mode == 0) {
          ((bf16_t*)outp)[(size_t)grow * DMODEL + gcol] = (bf16_t)v;
        } else if (mode == 1) {
          const int b = grow >> 11, n = grow & (SEQ - 1);
          const int h = gcol >> 6, d = gcol & (HDIM - 1);
          ((bf16_t*)outp)[(((size_t)b * NHEADS + h) * HDIM + d) * SEQ + n] = (bf16_t)v;
        } else {
          ((float*)outp)[(size_t)grow * DMODEL + gcol] = v + bias[gcol];
        }
      }
    }
  }
}

// ---------------- flash attention, one wave per 16 query rows ----------------
// Q,K: [b][n][h*64+d] bf16.   Vt: [b][h][d][n] bf16.   O: [b][n][h*64+d] bf16.
// K/V tiles (64x64 bf16) are double-buffered in LDS and streamed by the TDM,
// so softmax VALU co-executes with WMMA while the next tile is in flight.
__global__ void __launch_bounds__(32) attn_kernel(
    const bf16_t* __restrict__ Q, const bf16_t* __restrict__ K,
    const bf16_t* __restrict__ Vt, bf16_t* __restrict__ O) {
#if USE_TDM
  __shared__ bf16_t Ktile[2][64 * 64];  // [key][d]  8 KB per buffer
  __shared__ bf16_t Vtile[2][64 * 64];  // [d][key]  8 KB per buffer
#endif
  __shared__ bf16_t Plds[16 * 64];      // P-tile staging (C-layout -> A-layout)

  const int lane  = threadIdx.x;
  const int row16 = lane & 15;
  const int hf    = lane >> 4;
  const int qt = blockIdx.x, h = blockIdx.y, b = blockIdx.z;

  const size_t qkbase = (size_t)b * SEQ * DMODEL + (size_t)h * HDIM;
  const bf16_t* qrow  = Q + qkbase + (size_t)(qt * 16 + row16) * DMODEL;
  const bf16_t* kbase = K + qkbase;                                  // row stride DMODEL
  const bf16_t* vbase = Vt + ((size_t)b * NHEADS + h) * HDIM * SEQ;  // row stride SEQ

  // Q fragments resident for the whole row block (HD=64 => 2 K-chunks of 32)
  v16bf qa[2];
  qa[0] = load_a_frag(qrow, 0, hf);
  qa[1] = load_a_frag(qrow, 32, hf);

  v8f o[4];
#pragma unroll
  for (int ct = 0; ct < 4; ++ct) o[ct] = v8f_zero();
  float m[8], l[8];
#pragma unroll
  for (int r = 0; r < 8; ++r) { m[r] = -3.0e38f; l[r] = 0.0f; }

  const float scale = 0.125f;  // 1/sqrt(64)

#if USE_TDM
  const unsigned klds[2] = { (unsigned)(unsigned long long)&Ktile[0][0],
                             (unsigned)(unsigned long long)&Ktile[1][0] };
  const unsigned vlds[2] = { (unsigned)(unsigned long long)&Vtile[0][0],
                             (unsigned)(unsigned long long)&Vtile[1][0] };
  // prologue: stage tile 0
  tdm_load_2d(klds[0], kbase, 64, 64, DMODEL);
  tdm_load_2d(vlds[0], vbase, 64, 64, SEQ);
#endif

  for (int it = 0; it < SEQ / 64; ++it) {
    const int jt  = it * 64;
    const int buf = it & 1;
#if USE_TDM
    if (it + 1 < SEQ / 64) {  // stage next tile into the other buffer
      tdm_load_2d(klds[buf ^ 1], kbase + (size_t)(jt + 64) * DMODEL, 64, 64, DMODEL);
      tdm_load_2d(vlds[buf ^ 1], vbase + (jt + 64), 64, 64, SEQ);
      WAIT_TENSORCNT(2);  // current tile ready; next pair still in flight
    } else {
      WAIT_TENSORCNT(0);
    }
#endif
    // ---- S = Q * K^T for 64 keys (4 col-tiles x 2 K-chunks = 8 WMMAs) ----
    v8f s[4];
#pragma unroll
    for (int ct = 0; ct < 4; ++ct) {
      s[ct] = v8f_zero();
#pragma unroll
      for (int f = 0; f < 2; ++f) {
#if USE_TDM
        const bf16_t* kp = &Ktile[buf][(ct * 16 + row16) * 64 + f * 32 + hf * 16];
#else
        const bf16_t* kp =
            kbase + (size_t)(jt + ct * 16 + row16) * DMODEL + f * 32 + hf * 16;
#endif
        v16bf kf = *(const v16bf*)kp;
        s[ct]    = wmma_bf16(qa[f], kf, s[ct]);
      }
    }
    // ---- online softmax (row stats live in 16-lane halves) ----
    float tmax[8], tsum[8];
#pragma unroll
    for (int ct = 0; ct < 4; ++ct)
#pragma unroll
      for (int r = 0; r < 8; ++r) s[ct][r] *= scale;
#pragma unroll
    for (int r = 0; r < 8; ++r)
      tmax[r] = fmaxf(fmaxf(s[0][r], s[1][r]), fmaxf(s[2][r], s[3][r]));
#pragma unroll
    for (int mask = 1; mask <= 8; mask <<= 1)
#pragma unroll
      for (int r = 0; r < 8; ++r)
        tmax[r] = fmaxf(tmax[r], __shfl_xor(tmax[r], mask, 32));

    float alpha[8];
#pragma unroll
    for (int r = 0; r < 8; ++r) {
      const float mn = fmaxf(m[r], tmax[r]);
      alpha[r]       = __expf(m[r] - mn);
      m[r]           = mn;
      tsum[r]        = 0.0f;
    }
#pragma unroll
    for (int ct = 0; ct < 4; ++ct)
#pragma unroll
      for (int r = 0; r < 8; ++r) {
        const float p = __expf(s[ct][r] - m[r]);
        s[ct][r]      = p;
        tsum[r] += p;
      }
#pragma unroll
    for (int mask = 1; mask <= 8; mask <<= 1)
#pragma unroll
      for (int r = 0; r < 8; ++r) tsum[r] += __shfl_xor(tsum[r], mask, 32);
#pragma unroll
    for (int r = 0; r < 8; ++r) l[r] = l[r] * alpha[r] + tsum[r];
#pragma unroll
    for (int ct = 0; ct < 4; ++ct)
#pragma unroll
      for (int r = 0; r < 8; ++r) o[ct][r] *= alpha[r];

    // ---- P tile (C layout) -> LDS -> A fragments ----
    __syncthreads();
#pragma unroll
    for (int ct = 0; ct < 4; ++ct)
#pragma unroll
      for (int r = 0; r < 8; ++r)
        Plds[(hf * 8 + r) * 64 + ct * 16 + row16] = (bf16_t)s[ct][r];
    __syncthreads();

    // ---- O += P(16x64) * V(64x64)  (2 K-chunks x 4 col-tiles = 8 WMMAs) ----
#pragma unroll
    for (int kk = 0; kk < 2; ++kk) {
      v16bf pa = load_a_frag(Plds + (size_t)row16 * 64, kk * 32, hf);
#pragma unroll
      for (int ct = 0; ct < 4; ++ct) {
#if USE_TDM
        const bf16_t* vp = &Vtile[buf][(ct * 16 + row16) * 64 + kk * 32 + hf * 16];
#else
        const bf16_t* vp =
            vbase + (size_t)(ct * 16 + row16) * SEQ + jt + kk * 32 + hf * 16;
#endif
        v16bf vf = *(const v16bf*)vp;
        o[ct]    = wmma_bf16(pa, vf, o[ct]);
      }
    }
    __syncthreads();
  }

  // ---- finalize: O / l, store bf16 [b][n][h*64+d] ----
#pragma unroll
  for (int ct = 0; ct < 4; ++ct)
#pragma unroll
    for (int r = 0; r < 8; ++r) {
      const int grow = qt * 16 + hf * 8 + r;
      const int gcol = h * HDIM + ct * 16 + row16;
      O[((size_t)b * SEQ + grow) * DMODEL + gcol] = (bf16_t)(o[ct][r] / l[r]);
    }
}

// ---------------- host launch ----------------
extern "C" void kernel_launch(void* const* d_in, const int* in_sizes, int n_in,
                              void* d_out, int out_size, void* d_ws, size_t ws_size,
                              hipStream_t stream) {
  (void)in_sizes; (void)n_in; (void)out_size; (void)ws_size;
  const float* x  = (const float*)d_in[0];
  const float* Wq = (const float*)d_in[1];
  const float* Wk = (const float*)d_in[2];
  const float* Wv = (const float*)d_in[3];
  const float* Wo = (const float*)d_in[4];
  const float* bo = (const float*)d_in[5];

  const size_t nTok = (size_t)BATCH * SEQ;      // 8192
  const size_t nAct = nTok * DMODEL;            // 8,388,608
  const size_t nW   = (size_t)DMODEL * DMODEL;  // 1,048,576

  char* ws = (char*)d_ws;
  size_t off = 0;
  bf16_t* xb  = (bf16_t*)(ws + off); off += nAct * 2;
  bf16_t* wqb = (bf16_t*)(ws + off); off += nW * 2;
  bf16_t* wkb = (bf16_t*)(ws + off); off += nW * 2;
  bf16_t* wvb = (bf16_t*)(ws + off); off += nW * 2;
  bf16_t* wob = (bf16_t*)(ws + off); off += nW * 2;
  bf16_t* Qb  = (bf16_t*)(ws + off); off += nAct * 2;
  bf16_t* Kb  = (bf16_t*)(ws + off); off += nAct * 2;
  bf16_t* Vt  = (bf16_t*)(ws + off); off += nAct * 2;
  bf16_t* Ab  = (bf16_t*)(ws + off); off += nAct * 2;  // total ~92 MB

  // 1) f32 -> bf16 conversions
  cvt_f32_bf16_kernel<<<2048, 256, 0, stream>>>(x, xb, (int)(nAct / 4));
  cvt_f32_bf16_kernel<<<512, 256, 0, stream>>>(Wq, wqb, (int)(nW / 4));
  cvt_f32_bf16_kernel<<<512, 256, 0, stream>>>(Wk, wkb, (int)(nW / 4));
  cvt_f32_bf16_kernel<<<512, 256, 0, stream>>>(Wv, wvb, (int)(nW / 4));
  cvt_f32_bf16_kernel<<<512, 256, 0, stream>>>(Wo, wob, (int)(nW / 4));

  // 2) Q/K/V projections (V stored transposed per head)
  dim3 ggrid((unsigned)(nTok / 32), DMODEL / 64);
  gemm_xwT_kernel<<<ggrid, 32, 0, stream>>>(xb, wqb, nullptr, (void*)Qb, 0);
  gemm_xwT_kernel<<<ggrid, 32, 0, stream>>>(xb, wkb, nullptr, (void*)Kb, 0);
  gemm_xwT_kernel<<<ggrid, 32, 0, stream>>>(xb, wvb, nullptr, (void*)Vt, 1);

  // 3) flash attention (TDM-streamed K/V tiles)
  dim3 agrid(SEQ / 16, NHEADS, BATCH);
  attn_kernel<<<agrid, 32, 0, stream>>>(Qb, Kb, Vt, Ab);

  // 4) output projection (f32 + bias)
  gemm_xwT_kernel<<<ggrid, 32, 0, stream>>>(Ab, wob, bo, d_out, 2);
}
